// NKF_29240137351406
// MI455X (gfx1250) — matrix operand order repaired and checked by reading, not compiled, not run on hardware
//
#include <hip/hip_runtime.h>
#include <hip/hip_bf16.h>
#include <math.h>

typedef __attribute__((ext_vector_type(16))) _Float16 v16h;
typedef __attribute__((ext_vector_type(8)))  float    v8f;

#define NFFT   512
#define HOP    256
#define NSAMP  64000
#define BATCH  64
#define TFRM   251            // 1 + 64000/256
#define FBIN   257
#define BFROWS (BATCH * FBIN) // 16448 = 16 * 1028
#define LTAP   4
#define FCD    18
#define RDD    18
#define NTF    17             // ceil(257/16)
#define KCH    16             // 512/32
#define TABSZ  (KCH * NTF * 32 * 16)   // pre-swizzled B fragments
#define PLANE  ((size_t)BFROWS * (size_t)TFRM)
#define FTILES (BATCH * TFRM / 16)     // 1004 frame tiles per signal

// ---------------- WMMA helpers ----------------

__device__ __forceinline__ v8f wmma_f16(v16h a, v16h b, v8f c) {
  return __builtin_amdgcn_wmma_f32_16x16x32_f16(false, a, false, b, (short)0, c,
                                                false, false);
}

__device__ __forceinline__ v8f zero8() {
  v8f c = {0.f, 0.f, 0.f, 0.f, 0.f, 0.f, 0.f, 0.f};
  return c;
}

// A fragment (16x32 f16, row-major tile with given row stride).
// Layout: lane m=lane&15, hi=lane>>4; halves j0..15 -> K=16*hi+j
// (matches the B fragment order k = j + 16*(lane>>4)).
__device__ __forceinline__ v16h load_A(const _Float16* base, int rowStride, int kOff) {
  const int lane = threadIdx.x & 31;
  const int row  = lane & 15;
  const int hl   = lane >> 4;
  const _Float16* p = base + (size_t)row * rowStride + kOff + hl * 16;
  v16h a;
#pragma unroll
  for (int j = 0; j < 16; ++j) a[j] = p[j];
  return a;
}

// B fragment pre-swizzled as [lane][16] halves (contiguous per lane).
__device__ __forceinline__ v16h load_B(const _Float16* w) {
  const int lane = threadIdx.x & 31;
  const _Float16* p = w + lane * 16;
  v16h b;
#pragma unroll
  for (int j = 0; j < 16; ++j) b[j] = p[j];
  return b;
}

// C/D fragment dump: lane holds D[i+8*hi][lane&15 + 16*ntile], i=0..7.
__device__ __forceinline__ void dump_f32(float* buf, int ldb, v8f c, int ntile,
                                         const float* bias, int ncols) {
  const int lane = threadIdx.x & 31;
  const int n = ntile * 16 + (lane & 15);
  const int hl = lane >> 4;
  if (n < ncols) {
    const float bv = bias[n];
#pragma unroll
    for (int i = 0; i < 8; ++i) buf[(i + 8 * hl) * ldb + n] = c[i] + bv;
  }
}

__device__ __forceinline__ void dump_f16_bias(_Float16* buf, int ldb, v8f c, int ntile,
                                              const float* bias, int ncols) {
  const int lane = threadIdx.x & 31;
  const int n = ntile * 16 + (lane & 15);
  const int hl = lane >> 4;
  if (n < ncols) {
    const float bv = bias[n];
#pragma unroll
    for (int i = 0; i < 8; ++i) buf[(i + 8 * hl) * ldb + n] = (_Float16)(c[i] + bv);
  }
}

__device__ __forceinline__ void dump_f16_prelu(_Float16* buf, int ldb, v8f c, int ntile,
                                               const float* bias, int ncols, float alpha) {
  const int lane = threadIdx.x & 31;
  const int n = ntile * 16 + (lane & 15);
  const int hl = lane >> 4;
  if (n < ncols) {
    const float bv = bias[n];
#pragma unroll
    for (int i = 0; i < 8; ++i) {
      float v = c[i] + bv;
      v = (v >= 0.f) ? v : alpha * v;
      buf[(i + 8 * hl) * ldb + n] = (_Float16)v;
    }
  }
}

// ---------------- K0: windowed DFT tables, pre-swizzled B-fragment order ----------------
__global__ void nkf_dft_tables(_Float16* __restrict__ tabR, _Float16* __restrict__ tabI) {
  int idx = blockIdx.x * 256 + threadIdx.x;
  if (idx >= TABSZ) return;
  int j    = idx & 15;
  int lane = (idx >> 4) & 31;
  int nt   = (idx >> 9) % NTF;
  int kc   = idx / (NTF * 512);
  int n = nt * 16 + (lane & 15);
  int k = kc * 32 + (j + 16 * (lane >> 4));
  float vr = 0.f, vi = 0.f;
  if (n < FBIN) {
    const float twoPi = 6.283185307179586f;
    float w = 0.5f * (1.0f - cosf(twoPi * (float)k / (float)NFFT));  // periodic hann
    int ph = (k * n) & (NFFT - 1);                                   // exact mod-512 reduction
    float ang = twoPi * (float)ph / (float)NFFT;
    vr = w * cosf(ang);
    vi = -w * sinf(ang);
  }
  tabR[idx] = (_Float16)vr;
  tabI[idx] = (_Float16)vi;
}

// ---------------- K1: STFT as GEMM via WMMA; 4 waves/block share one frame tile ----------------
__global__ __launch_bounds__(128) void nkf_stft(const float* __restrict__ x,
                                                const float* __restrict__ y,
                                                const float* __restrict__ z,
                                                const _Float16* __restrict__ tabR,
                                                const _Float16* __restrict__ tabI,
                                                float* __restrict__ planes) {
  __shared__ _Float16 fr[16][NFFT];  // 16 reflect-padded frames, f16

  const int bid  = blockIdx.x;
  const int sig  = bid / FTILES;
  const int tile = bid - sig * FTILES;
  const float* s = (sig == 0) ? x : ((sig == 1) ? y : z);
  float* pr = planes + (size_t)sig * 2 * PLANE;
  float* pi = pr + PLANE;
  const int tid  = threadIdx.x;
  const int lane = tid & 31;
  const int wid  = tid >> 5;

  // Stage 16 frames (reflect padding, torch center=True semantics).
  for (int i = tid; i < 16 * NFFT; i += 128) {
    int m = i >> 9;
    int k = i & (NFFT - 1);
    int fidx = tile * 16 + m;
    int b = fidx / TFRM;
    int t = fidx - b * TFRM;
    int jj = t * HOP + k - (NFFT / 2);
    if (jj < 0) jj = -jj;
    if (jj >= NSAMP) jj = 2 * (NSAMP - 1) - jj;
    fr[m][k] = (_Float16)s[(size_t)b * NSAMP + jj];
  }
  __syncthreads();

  // Frequency N-tiles split across the 4 waves.
  for (int nt = wid; nt < NTF; nt += 4) {
    v8f cR = zero8();
    v8f cI = zero8();
    for (int kc = 0; kc < KCH; ++kc) {
      v16h a  = load_A(&fr[0][0], NFFT, kc * 32);
      v16h bR = load_B(tabR + ((size_t)(kc * NTF + nt) * 32) * 16);
      v16h bI = load_B(tabI + ((size_t)(kc * NTF + nt) * 32) * 16);
      cR = wmma_f16(a, bR, cR);
      cI = wmma_f16(a, bI, cI);
    }
    const int n  = nt * 16 + (lane & 15);
    const int hl = lane >> 4;
    if (n < FBIN) {
#pragma unroll
      for (int i = 0; i < 8; ++i) {
        int m = i + 8 * hl;
        int fidx = tile * 16 + m;
        int b = fidx / TFRM;
        int t = fidx - b * TFRM;
        size_t o = ((size_t)(b * FBIN + n)) * TFRM + t;
        pr[o] = cR[i];
        pi[o] = cI[i];
      }
    }
  }
}

// ---------------- K2: persistent WMMA scan; 4 waves/block, one GRU cell per wave ----------------

__device__ __forceinline__ void build_wfrag(_Float16* dst, const float* W, int Nrows,
                                            int Kcols, int ntiles, int tid) {
  // B layout: n = lane&15 (+16*tile), k = j + 16*(lane>>4); zero padding.
  for (int idx = tid; idx < ntiles * 512; idx += 128) {
    int tile = idx >> 9;
    int rem  = idx & 511;
    int lane = rem >> 4;
    int j    = rem & 15;
    int n = tile * 16 + (lane & 15);
    int k = j + 16 * (lane >> 4);
    float v = (n < Nrows && k < Kcols) ? W[n * Kcols + k] : 0.f;
    dst[idx] = (_Float16)v;
  }
}

__global__ __launch_bounds__(128) void nkf_scan(
    const float* __restrict__ planes, float* __restrict__ out,
    const float* __restrict__ Wr1, const float* __restrict__ br1,
    const float* __restrict__ Wi1, const float* __restrict__ bi1,
    const float* __restrict__ pa1,
    const float* __restrict__ Wih_r, const float* __restrict__ Whh_r,
    const float* __restrict__ bih_r, const float* __restrict__ bhh_r,
    const float* __restrict__ Wih_i, const float* __restrict__ Whh_i,
    const float* __restrict__ bih_i, const float* __restrict__ bhh_i,
    const float* __restrict__ Wr2, const float* __restrict__ br2,
    const float* __restrict__ Wi2, const float* __restrict__ bi2,
    const float* __restrict__ pa2,
    const float* __restrict__ Wr3, const float* __restrict__ br3,
    const float* __restrict__ Wi3, const float* __restrict__ bi3) {
  // Weight B fragments (shared, built once):
  // 0-1 Wr1 | 2-3 Wi1 | 4-7 Wih_r | 8-11 Whh_r | 12-15 Wih_i | 16-19 Whh_i
  // 20-21 Wr2 | 22-23 Wi2 | 24 Wr3 | 25 Wi3
  __shared__ _Float16 wfrag[26][32][16];
  // f16 A-staging tiles (K padded to 32, zero pad):
  // 0 featR 1 featI 2 aR 3 aI 4..7 h(rr,ir,ri,ii) 8 gR 9 gI 10 oR 11 oI
  __shared__ _Float16 stage[12][16][32];
  __shared__ _Float16 giB[4][16][64];   // per-cell input-gate pre-activations (+bias)
  __shared__ _Float16 ghB[4][16][64];   // per-cell hidden-gate pre-activations (+bias)
  __shared__ float hstate[4][16][RDD];
  __shared__ float hprR[16][LTAP], hprI[16][LTAP];
  __shared__ float hpoR[16][LTAP], hpoI[16][LTAP];
  __shared__ float xwR[16][LTAP], xwI[16][LTAP];
  __shared__ float kgR[16][LTAP], kgI[16][LTAP];
  __shared__ float eR[16], eI[16], yRe[16], yIm[16];

  const int tid  = threadIdx.x;
  const int lane = tid & 31;
  const int wid  = tid >> 5;          // wave id 0..3 == GRU cell id
  const int rowBase = blockIdx.x * 16;
  const float a1 = pa1[0];
  const float a2 = pa2[0];

  const float* Xr = planes;
  const float* Xi = planes + PLANE;
  const float* Yr = planes + 2 * PLANE;
  const float* Yi = planes + 3 * PLANE;
  const float* Zr = planes + 4 * PLANE;
  const float* Zi = planes + 5 * PLANE;

  // Build weight fragments (whole block cooperates).
  build_wfrag(&wfrag[0][0][0],  Wr1,   FCD,     2 * LTAP + 1, 2, tid);
  build_wfrag(&wfrag[2][0][0],  Wi1,   FCD,     2 * LTAP + 1, 2, tid);
  build_wfrag(&wfrag[4][0][0],  Wih_r, 3 * RDD, FCD,          4, tid);
  build_wfrag(&wfrag[8][0][0],  Whh_r, 3 * RDD, RDD,          4, tid);
  build_wfrag(&wfrag[12][0][0], Wih_i, 3 * RDD, FCD,          4, tid);
  build_wfrag(&wfrag[16][0][0], Whh_i, 3 * RDD, RDD,          4, tid);
  build_wfrag(&wfrag[20][0][0], Wr2,   FCD,     RDD,          2, tid);
  build_wfrag(&wfrag[22][0][0], Wi2,   FCD,     RDD,          2, tid);
  build_wfrag(&wfrag[24][0][0], Wr3,   LTAP,    FCD,          1, tid);
  build_wfrag(&wfrag[25][0][0], Wi3,   LTAP,    FCD,          1, tid);

  // Zero staging + state.
  for (int i = tid; i < 12 * 16 * 32; i += 128) (&stage[0][0][0])[i] = (_Float16)0.f;
  for (int i = tid; i < 4 * 16 * RDD; i += 128) (&hstate[0][0][0])[i] = 0.f;
  if (tid < 16) {
#pragma unroll
    for (int l = 0; l < LTAP; ++l) {
      hprR[tid][l] = 0.f; hprI[tid][l] = 0.f;
      hpoR[tid][l] = 0.f; hpoI[tid][l] = 0.f;
      xwR[tid][l] = 0.f;  xwI[tid][l] = 0.f;
    }
  }
  __syncthreads();

  for (int t = 0; t < TFRM; ++t) {
    // ---- Phase 1: stream inputs, build feature tile (threads 0..15 = rows) ----
    if (tid < 16) {
      const size_t rt = (size_t)(rowBase + tid) * TFRM + t;
      const float xr = Xr[rt], xi = Xi[rt];
      const float yr = Yr[rt], yi = Yi[rt];
      const float zr = Zr[rt], zi = Zi[rt];
#pragma unroll
      for (int l = 0; l < LTAP - 1; ++l) { xwR[tid][l] = xwR[tid][l + 1]; xwI[tid][l] = xwI[tid][l + 1]; }
      xwR[tid][LTAP - 1] = xr; xwI[tid][LTAP - 1] = xi;
      const float er = yr - zr, ei = yi - zi;
      eR[tid] = er; eI[tid] = ei; yRe[tid] = yr; yIm[tid] = yi;
      _Float16* fR = &stage[0][tid][0];
      _Float16* fI = &stage[1][tid][0];
#pragma unroll
      for (int l = 0; l < LTAP; ++l) { fR[l] = (_Float16)xwR[tid][l]; fI[l] = (_Float16)xwI[tid][l]; }
      fR[LTAP] = (_Float16)er; fI[LTAP] = (_Float16)ei;
#pragma unroll
      for (int l = 0; l < LTAP; ++l) {
        fR[LTAP + 1 + l] = (_Float16)(hpoR[tid][l] - hprR[tid][l]);
        fI[LTAP + 1 + l] = (_Float16)(hpoI[tid][l] - hprI[tid][l]);
        hprR[tid][l] = hpoR[tid][l];   // h_prior = h_post
        hprI[tid][l] = hpoI[tid][l];
      }
    }
    __syncthreads();

    // ---- fc_in: ComplexDense(9->18)+PReLU; one WMMA per wave ----
    {
      v16h fA = load_A(&stage[(wid < 2) ? 0 : 1][0][0], 32, 0);
      v8f c = wmma_f16(fA, load_B(&wfrag[wid][0][0]), zero8());
      dump_f16_prelu(&stage[(wid < 2) ? 2 : 3][0][0], 32, c, wid & 1,
                     (wid < 2) ? br1 : bi1, FCD, a1);
    }
    __syncthreads();

    // ---- 4 GRU cells in parallel: wave w owns cell w ----
    // cells: 0=Frr(aR,Wr) 1=Fir(aI,Wr) 2=Fri(aR,Wi) 3=Fii(aI,Wi)
    {
      const int cell = wid;
      const int aIdx = (cell == 0 || cell == 2) ? 2 : 3;
      const int wIh  = (cell < 2) ? 4 : 12;
      const int wHh  = (cell < 2) ? 8 : 16;
      const float* bih = (cell < 2) ? bih_r : bih_i;
      const float* bhh = (cell < 2) ? bhh_r : bhh_i;
      v16h Af = load_A(&stage[aIdx][0][0], 32, 0);
      v16h Hf = load_A(&stage[4 + cell][0][0], 32, 0);
#pragma unroll
      for (int nt = 0; nt < 4; ++nt) {
        v8f c0 = wmma_f16(Af, load_B(&wfrag[wIh + nt][0][0]), zero8());
        dump_f16_bias(&giB[cell][0][0], 64, c0, nt, bih, 3 * RDD);
        v8f c1 = wmma_f16(Hf, load_B(&wfrag[wHh + nt][0][0]), zero8());
        dump_f16_bias(&ghB[cell][0][0], 64, c1, nt, bhh, 3 * RDD);
      }
      __syncthreads();
      // Gate math: each wave handles its own cell, 9 elements/lane (32*9 = 16*18).
      const int base = lane * 9;
#pragma unroll
      for (int m = 0; m < 9; ++m) {
        const int idx = base + m;
        const int row = idx / RDD;
        const int col = idx - row * RDD;
        const float ir = (float)giB[cell][row][col];
        const float iz = (float)giB[cell][row][col + RDD];
        const float ig = (float)giB[cell][row][col + 2 * RDD];
        const float hr = (float)ghB[cell][row][col];
        const float hz = (float)ghB[cell][row][col + RDD];
        const float hg = (float)ghB[cell][row][col + 2 * RDD];
        const float r  = 1.f / (1.f + __expf(-(ir + hr)));
        const float zg = 1.f / (1.f + __expf(-(iz + hz)));
        const float nn = tanhf(ig + r * hg);
        const float ho = hstate[cell][row][col];
        const float hn = (1.f - zg) * nn + zg * ho;
        hstate[cell][row][col] = hn;
        stage[4 + cell][row][col] = (_Float16)hn;
      }
    }
    __syncthreads();

    // ---- complex combine: gr = Frr - Fii, gi = Fri + Fir (whole block) ----
    for (int i = tid; i < 16 * RDD; i += 128) {
      const int row = i / RDD;
      const int col = i - row * RDD;
      stage[8][row][col] = (_Float16)(hstate[0][row][col] - hstate[3][row][col]);
      stage[9][row][col] = (_Float16)(hstate[2][row][col] + hstate[1][row][col]);
    }
    __syncthreads();

    // ---- fc_out stage 1: Dense(18->18)+PReLU; one WMMA per wave ----
    {
      v16h gA = load_A(&stage[(wid < 2) ? 8 : 9][0][0], 32, 0);
      v8f c = wmma_f16(gA, load_B(&wfrag[20 + wid][0][0]), zero8());
      dump_f16_prelu(&stage[(wid < 2) ? 10 : 11][0][0], 32, c, wid & 1,
                     (wid < 2) ? br2 : bi2, FCD, a2);
    }
    __syncthreads();

    // ---- fc_out stage 2: Dense(18->4) => Kalman gain; waves 0,1 ----
    if (wid < 2) {
      v16h oA = load_A(&stage[10 + wid][0][0], 32, 0);
      v8f c = wmma_f16(oA, load_B(&wfrag[24 + wid][0][0]), zero8());
      dump_f32((wid == 0) ? &kgR[0][0] : &kgI[0][0], LTAP, c, 0,
               (wid == 0) ? br3 : bi3, LTAP);
    }
    __syncthreads();

    // ---- Phase 3: Kalman update, echo, output (threads 0..15 = rows) ----
    if (tid < 16) {
      const float er = eR[tid], ei = eI[tid];
      float echoR = 0.f, echoI = 0.f;
#pragma unroll
      for (int l = 0; l < LTAP; ++l) {
        const float kr = kgR[tid][l], ki = kgI[tid][l];
        float hr = hprR[tid][l] + kr * er - ki * ei;
        float hi = hprI[tid][l] + kr * ei + ki * er;
        hr = fminf(5.f, fmaxf(-5.f, hr));
        hi = fminf(5.f, fmaxf(-5.f, hi));
        hpoR[tid][l] = hr; hpoI[tid][l] = hi;
        echoR += xwR[tid][l] * hr - xwI[tid][l] * hi;
        echoI += xwR[tid][l] * hi + xwI[tid][l] * hr;
      }
      const size_t rt = (size_t)(rowBase + tid) * TFRM + t;
      out[2 * rt + 0] = yRe[tid] - echoR;
      out[2 * rt + 1] = yIm[tid] - echoI;
    }
    __syncthreads();
  }
}

// ---------------- host launcher ----------------

extern "C" void kernel_launch(void* const* d_in, const int* in_sizes, int n_in,
                              void* d_out, int out_size, void* d_ws, size_t ws_size,
                              hipStream_t stream) {
  (void)in_sizes; (void)n_in; (void)out_size; (void)ws_size;
  const float* x     = (const float*)d_in[0];
  const float* y     = (const float*)d_in[1];
  const float* z     = (const float*)d_in[2];
  const float* Wr1   = (const float*)d_in[3];
  const float* br1   = (const float*)d_in[4];
  const float* Wi1   = (const float*)d_in[5];
  const float* bi1   = (const float*)d_in[6];
  const float* a1    = (const float*)d_in[7];
  const float* Wih_r = (const float*)d_in[8];
  const float* Whh_r = (const float*)d_in[9];
  const float* bih_r = (const float*)d_in[10];
  const float* bhh_r = (const float*)d_in[11];
  const float* Wih_i = (const float*)d_in[12];
  const float* Whh_i = (const float*)d_in[13];
  const float* bih_i = (const float*)d_in[14];
  const float* bhh_i = (const float*)d_in[15];
  const float* Wr2   = (const float*)d_in[16];
  const float* br2   = (const float*)d_in[17];
  const float* Wi2   = (const float*)d_in[18];
  const float* bi2   = (const float*)d_in[19];
  const float* a2    = (const float*)d_in[20];
  const float* Wr3   = (const float*)d_in[21];
  const float* br3   = (const float*)d_in[22];
  const float* Wi3   = (const float*)d_in[23];
  const float* bi3   = (const float*)d_in[24];

  char* ws = (char*)d_ws;
  _Float16* tabR = (_Float16*)ws;
  _Float16* tabI = tabR + TABSZ;
  float* planes  = (float*)(ws + 2 * (size_t)TABSZ * sizeof(_Float16));

  nkf_dft_tables<<<dim3((TABSZ + 255) / 256), dim3(256), 0, stream>>>(tabR, tabI);
  nkf_stft<<<dim3(3 * FTILES), dim3(128), 0, stream>>>(x, y, z, tabR, tabI, planes);
  nkf_scan<<<dim3(BFROWS / 16), dim3(128), 0, stream>>>(
      planes, (float*)d_out,
      Wr1, br1, Wi1, bi1, a1,
      Wih_r, Whh_r, bih_r, bhh_r,
      Wih_i, Whh_i, bih_i, bhh_i,
      Wr2, br2, Wi2, bi2, a2,
      Wr3, br3, Wi3, bi3);
}
